// Decoder_31181462569441
// MI455X (gfx1250) — matrix operand order repaired and checked by reading, not compiled, not run on hardware
//
#include <hip/hip_runtime.h>
#include <hip/hip_bf16.h>

// ---------------------------------------------------------------------------
// Problem constants (match reference)
// ---------------------------------------------------------------------------
#define BB 64      // batch
#define SS 512     // encoder seq len
#define HH 1024    // hidden
#define EE 10      // n_events
#define N4H 4096   // 4*H
#define LSTM_IN 1035  // H + E + 1

typedef __bf16 bf16_t;
typedef __attribute__((ext_vector_type(16))) __bf16 v16bf;
typedef __attribute__((ext_vector_type(8)))  float  v8f;
typedef __attribute__((ext_vector_type(4)))  int    i32x4;

#define AS1 __attribute__((address_space(1)))
#define AS3 __attribute__((address_space(3)))

// ---- CDNA5 native transcendental tanh (V_TANH_F32) ------------------------
#if __has_builtin(__builtin_amdgcn_tanhf)
__device__ __forceinline__ float fast_tanh(float x) {
  return __builtin_amdgcn_tanhf(x);
}
#else
#warning "gfx1250 native tanh builtin not available; using libm tanhf"
__device__ __forceinline__ float fast_tanh(float x) { return tanhf(x); }
#endif

__device__ __forceinline__ float sigmoidf_fast(float x) {
  // sigmoid(x) = 0.5*tanh(x/2) + 0.5  -> single native trans op
  return 0.5f * fast_tanh(0.5f * x) + 0.5f;
}

// ---- CDNA5 async global->LDS DMA path --------------------------------------
#if __has_builtin(__builtin_amdgcn_global_load_async_to_lds_b128) && \
    __has_builtin(__builtin_amdgcn_s_wait_asynccnt)
#define USE_ASYNC_LDS 1
#else
#define USE_ASYNC_LDS 0
#warning "gfx1250 async-to-LDS builtins not available; context kernel uses direct loads"
#endif

#if USE_ASYNC_LDS
// The builtin takes int4 pointers in AS1 (global) / AS3 (LDS). Build them via
// integers: a global VA is numerically the AS1 address; a generic LDS
// pointer's low 32 bits are the AS3 offset.
__device__ __forceinline__ AS1 i32x4* as_global(const void* p) {
  return (AS1 i32x4*)(unsigned long long)(uintptr_t)p;
}
__device__ __forceinline__ AS3 i32x4* as_lds(const void* p) {
  return (AS3 i32x4*)(unsigned)(uintptr_t)p;
}
#endif

__device__ __forceinline__ v8f wmma_bf16(v16bf a, v16bf b, v8f c) {
  // D = A(16x32 bf16) x B(32x16 bf16) + C(16x16 f32)
  return __builtin_amdgcn_wmma_f32_16x16x32_bf16(
      /*neg_a=*/false, a, /*neg_b=*/false, b,
      /*c_mod=*/(short)0, c, /*reuse_a=*/false, /*reuse_b=*/false);
}

// A fragment: 16x32 bf16 tile from an LDS row-major [*, ldm] bf16 array.
// ISA layout: lanes 0-15 -> M=lane, K base 0; lanes 16-31 -> M=lane-16, K base 8.
// v[0..7] = K kb..kb+7 ; v[8..15] = K kb+16..kb+23.
__device__ __forceinline__ v16bf load_a_frag(const bf16_t* at, int ldm,
                                             int m0, int k0, int lane) {
  const int m  = m0 + (lane & 15);
  const int kb = k0 + ((lane >> 4) << 3);
  const bf16_t* p = at + m * ldm + kb;
  v16bf v;
#pragma unroll
  for (int j = 0; j < 8; ++j) { v[j] = p[j]; v[8 + j] = p[16 + j]; }
  return v;
}

// B fragment: 32x16 bf16 with B[k][n] = W[n0+n][k0+k], W row-major f32 [*, ldk].
// lanes 0-15: K 0..7,16..23, lanes 16-31: K 8..15,24..31 (column n = lane&15).
__device__ __forceinline__ v16bf load_b_frag(const float* __restrict__ W, int ldk,
                                             int n0, int k0, int lane) {
  const int n  = n0 + (lane & 15);
  const int kb = k0 + ((lane >> 4) << 3);
  const float* p = W + (size_t)n * ldk + kb;
  v16bf v;
#pragma unroll
  for (int j = 0; j < 8; ++j) { v[j] = (bf16_t)p[j]; v[8 + j] = (bf16_t)p[16 + j]; }
  return v;
}

// ---------------------------------------------------------------------------
// 1) decproj[b,g] = sum_h h0[b,h] * W_dec[g,h]     (64x1024, K=1024)
// ---------------------------------------------------------------------------
__global__ __launch_bounds__(256) void decproj_kernel(
    const float* __restrict__ h0, const float* __restrict__ W_dec,
    float* __restrict__ decproj) {
  const int b = blockIdx.y;
  const int g = blockIdx.x * 256 + threadIdx.x;
  const float* hb = h0 + (size_t)b * HH;
  const float* wr = W_dec + (size_t)g * HH;
  float acc = 0.f;
#pragma unroll 4
  for (int h = 0; h < HH; ++h) acc = fmaf(hb[h], wr[h], acc);
  decproj[(size_t)b * HH + g] = acc;
}

// ---------------------------------------------------------------------------
// 2) Fused attention-score GEMM (the 68.7 GFLOP piece):
//    score[b,s] = sum_g w_val[g] * tanh( enc[b,s,:]·W_enc[g,:] + decproj[b,g] )
//    Block: 64 rows (s) of one batch, A tile staged bf16 in 128KB dynamic LDS.
//    8 waves x (8 n-chunks x 32 k-steps x 4 m-tiles) WMMA_F32_16X16X32_BF16.
// ---------------------------------------------------------------------------
#define AMT 64   // rows per block
__global__ __launch_bounds__(256) void attn_score_kernel(
    const float* __restrict__ enc,      // [B,S,H]
    const float* __restrict__ W_enc,    // [H,H]  (rows g, cols h)
    const float* __restrict__ decproj,  // [B,H]
    const float* __restrict__ w_val,    // [H]
    float* __restrict__ scores) {       // [B,S]
  extern __shared__ char smem[];
  bf16_t* a_tile   = (bf16_t*)smem;                               // [AMT][HH]
  float*  s_scores = (float*)(smem + (size_t)AMT * HH * sizeof(bf16_t)); // [AMT]

  const int b    = blockIdx.y;
  const int s0   = blockIdx.x * AMT;
  const int tid  = threadIdx.x;
  const int wave = tid >> 5;
  const int lane = tid & 31;

  // Stage A (enc rows) as bf16 into LDS; zero the score accumulators.
  const float* encb = enc + ((size_t)b * SS + s0) * HH;
  for (int i = tid; i < AMT * HH; i += 256) a_tile[i] = (bf16_t)encb[i];
  for (int m = tid; m < AMT; m += 256) s_scores[m] = 0.f;
  __syncthreads();

  float spart[4][8];
#pragma unroll
  for (int mt = 0; mt < 4; ++mt)
#pragma unroll
    for (int r = 0; r < 8; ++r) spart[mt][r] = 0.f;

  const int ncol = lane & 15;

  for (int j = 0; j < 8; ++j) {           // n-chunk: this wave's gate columns
    const int n0 = wave * 16 + j * 128;
    v8f acc[4] = {v8f{}, v8f{}, v8f{}, v8f{}};
    for (int ks = 0; ks < HH; ks += 32) { // full K before the tanh epilogue
      v16bf bf = load_b_frag(W_enc, HH, n0, ks, lane);
#pragma unroll
      for (int mt = 0; mt < 4; ++mt) {
        v16bf af = load_a_frag(a_tile, HH, mt * 16, ks, lane);
        acc[mt] = wmma_bf16(af, bf, acc[mt]);
      }
    }
    // epilogue: tanh(+decproj) * w_val, accumulate row partials in registers
    const int n = n0 + ncol;
    const float dp = decproj[(size_t)b * HH + n];
    const float wv = w_val[n];
#pragma unroll
    for (int mt = 0; mt < 4; ++mt)
#pragma unroll
      for (int r = 0; r < 8; ++r)
        spart[mt][r] += fast_tanh(acc[mt][r] + dp) * wv;
  }

  // Reduce the 16 columns held across lanes (xor-shuffle within 16-lane halves),
  // then lanes 0 / 16 fold their 8 rows into LDS.
#pragma unroll
  for (int mt = 0; mt < 4; ++mt)
#pragma unroll
    for (int r = 0; r < 8; ++r) {
      float v = spart[mt][r];
      v += __shfl_xor(v, 1);
      v += __shfl_xor(v, 2);
      v += __shfl_xor(v, 4);
      v += __shfl_xor(v, 8);
      if ((lane & 15) == 0)
        atomicAdd(&s_scores[mt * 16 + ((lane >> 4) << 3) + r], v);
    }
  __syncthreads();

  for (int m = tid; m < AMT; m += 256)
    scores[(size_t)b * SS + s0 + m] = s_scores[m];
}

// ---------------------------------------------------------------------------
// 3) softmax over S per batch -> attn_weight (written straight into d_out)
// ---------------------------------------------------------------------------
__global__ __launch_bounds__(256) void softmax_kernel(
    const float* __restrict__ scores, float* __restrict__ wout) {
  __shared__ float sv[SS];
  __shared__ float red[256];
  const int b = blockIdx.x, t = threadIdx.x;
  sv[t]       = scores[(size_t)b * SS + t];
  sv[t + 256] = scores[(size_t)b * SS + t + 256];
  __syncthreads();
  float m = fmaxf(sv[t], sv[t + 256]);
  red[t] = m; __syncthreads();
  for (int o = 128; o > 0; o >>= 1) {
    if (t < o) red[t] = fmaxf(red[t], red[t + o]);
    __syncthreads();
  }
  m = red[0];
  __syncthreads();
  const float e0 = __expf(sv[t] - m), e1 = __expf(sv[t + 256] - m);
  red[t] = e0 + e1; __syncthreads();
  for (int o = 128; o > 0; o >>= 1) {
    if (t < o) red[t] += red[t + o];
    __syncthreads();
  }
  const float inv = 1.0f / red[0];
  wout[(size_t)b * SS + t]       = e0 * inv;
  wout[(size_t)b * SS + t + 256] = e1 * inv;
}

// ---------------------------------------------------------------------------
// 4) context[b,h] = sum_s w[b,s] * enc[b,s,h]   (second 134MB pass, BW-bound)
//    Double-buffered GLOBAL_LOAD_ASYNC_TO_LDS_B128 pipeline; each tile is
//    8 seq rows x 256 h columns (8KB), prefetched while the previous tile is
//    reduced. Synchronized with s_wait_asynccnt + workgroup barrier.
// ---------------------------------------------------------------------------
#define CT 8   // seq rows per async tile
__global__ __launch_bounds__(256) void context_kernel(
    const float* __restrict__ enc, const float* __restrict__ w,
    float* __restrict__ ctx) {
  __shared__ float wsm[SS];
  const int b = blockIdx.y, t = threadIdx.x;
  const int h0 = blockIdx.x * 256;
  wsm[t]       = w[(size_t)b * SS + t];
  wsm[t + 256] = w[(size_t)b * SS + t + 256];

#if USE_ASYNC_LDS
  __shared__ float buf[2][CT * 256];
  const float* base = enc + (size_t)b * SS * HH + h0;
  // One tile = CT*256 floats = CT*64 16B chunks = 512 chunks; 2 per thread.
  auto stage = [&](int bi, int tile) {
#pragma unroll
    for (int q = 0; q < 2; ++q) {
      const int i = t + q * 256;          // chunk index
      const int r = i >> 6;               // row within tile
      const int c = (i & 63) * 4;         // float column within row
      const float* g = base + (size_t)(tile * CT + r) * HH + c;
      __builtin_amdgcn_global_load_async_to_lds_b128(
          as_global(g), as_lds(&buf[bi][r * 256 + c]), 0, 0);
    }
  };
  float acc = 0.f;
  stage(0, 0);
  __syncthreads();  // covers wsm staging
  for (int tile = 0; tile < SS / CT; ++tile) {
    const int cur = tile & 1;
    if (tile + 1 < SS / CT) {
      stage(cur ^ 1, tile + 1);
      __builtin_amdgcn_s_wait_asynccnt(2);  // current tile done, next in flight
    } else {
      __builtin_amdgcn_s_wait_asynccnt(0);
    }
    __syncthreads();                        // all waves' DMA landed
#pragma unroll
    for (int r = 0; r < CT; ++r)
      acc = fmaf(wsm[tile * CT + r], buf[cur][r * 256 + t], acc);
    __syncthreads();                        // done reading before overwrite
  }
  ctx[(size_t)b * HH + h0 + t] = acc;
#else
  __syncthreads();
  const float* e = enc + (size_t)b * SS * HH + h0 + t;
  float acc = 0.f;
#pragma unroll 4
  for (int s = 0; s < SS; ++s) acc = fmaf(wsm[s], e[(size_t)s * HH], acc);
  ctx[(size_t)b * HH + h0 + t] = acc;
#endif
}

// ---------------------------------------------------------------------------
// 5) sim[b,:] = (ctx[b,:] @ events^T) @ W_gate^T   (tiny)
// ---------------------------------------------------------------------------
__global__ __launch_bounds__(64) void sim_kernel(
    const float* __restrict__ ctx, const float* __restrict__ em,
    const float* __restrict__ Wg, float* __restrict__ sim) {
  __shared__ float t1[EE];
  const int b = blockIdx.x, t = threadIdx.x;
  if (t < EE) t1[t] = 0.f;
  __syncthreads();
  for (int e = 0; e < EE; ++e) {
    float p = 0.f;
    for (int h = t; h < HH; h += 64)
      p = fmaf(ctx[(size_t)b * HH + h], em[(size_t)e * HH + h], p);
    atomicAdd(&t1[e], p);
  }
  __syncthreads();
  if (t < EE) {
    float s = 0.f;
#pragma unroll
    for (int e = 0; e < EE; ++e) s = fmaf(t1[e], Wg[t * EE + e], s);
    sim[(size_t)b * EE + t] = s;
  }
}

// ---------------------------------------------------------------------------
// 6) LSTM pre-activations via WMMA:
//    gates[b,n] = ctx·W_ih[n,0:1024] + h0·W_hh[n,:] + b_ih[n]+b_hh[n]
//               + sum_e sim[b,e]·W_ih[n,1024+e] + x[b]·W_ih[n,1034]
// ---------------------------------------------------------------------------
__global__ __launch_bounds__(256) void gates_kernel(
    const float* __restrict__ ctx,  const float* __restrict__ h0,
    const float* __restrict__ sim,  const float* __restrict__ x,
    const float* __restrict__ W_ih, const float* __restrict__ W_hh,
    const float* __restrict__ b_ih, const float* __restrict__ b_hh,
    float* __restrict__ gates) {
  __shared__ bf16_t a1[BB * 128];   // ctx slab
  __shared__ bf16_t a2[BB * 128];   // h0 slab
  __shared__ float  sim_s[BB * EE];
  __shared__ float  x_s[BB];

  const int tid  = threadIdx.x;
  const int wave = tid >> 5;
  const int lane = tid & 31;
  const int n0   = blockIdx.x * 128 + wave * 16;

  for (int i = tid; i < BB * EE; i += 256) sim_s[i] = sim[i];
  if (tid < BB) x_s[tid] = x[tid];

  v8f acc[4] = {v8f{}, v8f{}, v8f{}, v8f{}};
  for (int slab = 0; slab < 8; ++slab) {
    __syncthreads();
    const int k0 = slab * 128;
    for (int i = tid; i < BB * 128; i += 256) {
      const int r = i >> 7, c = i & 127;
      a1[i] = (bf16_t)ctx[(size_t)r * HH + k0 + c];
      a2[i] = (bf16_t)h0[(size_t)r * HH + k0 + c];
    }
    __syncthreads();
    for (int ks = 0; ks < 128; ks += 32) {
      v16bf b1 = load_b_frag(W_ih, LSTM_IN, n0, k0 + ks, lane);
      v16bf b2 = load_b_frag(W_hh, HH,      n0, k0 + ks, lane);
#pragma unroll
      for (int mt = 0; mt < 4; ++mt) {
        v16bf af1 = load_a_frag(a1, 128, mt * 16, ks, lane);
        acc[mt] = wmma_bf16(af1, b1, acc[mt]);
        v16bf af2 = load_a_frag(a2, 128, mt * 16, ks, lane);
        acc[mt] = wmma_bf16(af2, b2, acc[mt]);
      }
    }
  }

  // Epilogue: biases + sim/x remainder columns of W_ih, then store.
  const int n = n0 + (lane & 15);
  const float bias = b_ih[n] + b_hh[n];
  float wex[EE + 1];
#pragma unroll
  for (int e = 0; e <= EE; ++e) wex[e] = W_ih[(size_t)n * LSTM_IN + HH + e];
#pragma unroll
  for (int mt = 0; mt < 4; ++mt)
#pragma unroll
    for (int r = 0; r < 8; ++r) {
      const int row = mt * 16 + ((lane >> 4) << 3) + r;  // batch index
      float extra = x_s[row] * wex[EE];
#pragma unroll
      for (int e = 0; e < EE; ++e) extra = fmaf(sim_s[row * EE + e], wex[e], extra);
      gates[(size_t)row * N4H + n] = acc[mt][r] + bias + extra;
    }
}

// ---------------------------------------------------------------------------
// 7) LSTM cell update + LayerNorm + final linear. One block per batch row.
// ---------------------------------------------------------------------------
__global__ __launch_bounds__(256) void lstm_ln_kernel(
    const float* __restrict__ gates, const float* __restrict__ c0,
    const float* __restrict__ ln_g,  const float* __restrict__ ln_b,
    const float* __restrict__ W_fin, const float* __restrict__ b_fin,
    float* __restrict__ out_fin, float* __restrict__ out_h,
    float* __restrict__ out_c) {
  __shared__ float red[256];
  __shared__ float red2[256];
  const int b = blockIdx.x, t = threadIdx.x;
  const float* g = gates + (size_t)b * N4H;

  float hn[4], sum = 0.f, sumsq = 0.f;
#pragma unroll
  for (int j = 0; j < 4; ++j) {
    const int h = t + j * 256;
    const float si = sigmoidf_fast(g[h]);
    const float sf = sigmoidf_fast(g[HH + h]);
    const float sg = fast_tanh(g[2 * HH + h]);
    const float so = sigmoidf_fast(g[3 * HH + h]);
    const float c  = sf * c0[(size_t)b * HH + h] + si * sg;
    const float hh = so * fast_tanh(c);
    hn[j] = hh;
    sum += hh; sumsq += hh * hh;
    out_h[(size_t)b * HH + h] = hh;
    out_c[(size_t)b * HH + h] = c;
  }
  red[t] = sum; red2[t] = sumsq;
  __syncthreads();
  for (int o = 128; o > 0; o >>= 1) {
    if (t < o) { red[t] += red[t + o]; red2[t] += red2[t + o]; }
    __syncthreads();
  }
  const float mu   = red[0] * (1.0f / HH);
  const float var  = red2[0] * (1.0f / HH) - mu * mu;
  const float rstd = rsqrtf(var + 1e-5f);
  __syncthreads();

  float fin = 0.f;
#pragma unroll
  for (int j = 0; j < 4; ++j) {
    const int h = t + j * 256;
    const float y = (hn[j] - mu) * rstd * ln_g[h] + ln_b[h];
    fin = fmaf(y, W_fin[h], fin);
  }
  red[t] = fin;
  __syncthreads();
  for (int o = 128; o > 0; o >>= 1) {
    if (t < o) red[t] += red[t + o];
    __syncthreads();
  }
  if (t == 0) out_fin[b] = red[0] + b_fin[0];
}

// ---------------------------------------------------------------------------
// Launch
// ---------------------------------------------------------------------------
extern "C" void kernel_launch(void* const* d_in, const int* in_sizes, int n_in,
                              void* d_out, int out_size, void* d_ws, size_t ws_size,
                              hipStream_t stream) {
  const float* x       = (const float*)d_in[0];
  const float* h0      = (const float*)d_in[1];   // [1,B,H]
  const float* c0      = (const float*)d_in[2];
  const float* enc     = (const float*)d_in[3];   // [B,S,H]
  const float* W_enc   = (const float*)d_in[4];
  const float* W_dec   = (const float*)d_in[5];
  const float* w_val   = (const float*)d_in[6];
  const float* W_gate  = (const float*)d_in[7];
  const float* events  = (const float*)d_in[8];
  const float* W_ih    = (const float*)d_in[9];
  const float* W_hh    = (const float*)d_in[10];
  const float* b_ih    = (const float*)d_in[11];
  const float* b_hh    = (const float*)d_in[12];
  const float* ln_g    = (const float*)d_in[13];
  const float* ln_b    = (const float*)d_in[14];
  const float* W_fin   = (const float*)d_in[15];
  const float* b_fin   = (const float*)d_in[16];

  // workspace layout (floats)
  float* ws       = (float*)d_ws;
  float* decproj  = ws;                   // [B,H]      65536
  float* scores   = ws + 65536;           // [B,S]      32768
  float* context  = ws + 98304;           // [B,H]      65536
  float* gates    = ws + 163840;          // [B,4H]     262144

  // output layout: fin(64) | h_new(65536) | c_new(65536) | attn_w(32768) | sim(640)
  float* out      = (float*)d_out;
  float* out_fin  = out;
  float* out_h    = out + 64;
  float* out_c    = out + 64 + BB * HH;
  float* out_attn = out + 64 + 2 * BB * HH;
  float* out_sim  = out + 64 + 2 * BB * HH + BB * SS;

  decproj_kernel<<<dim3(HH / 256, BB), 256, 0, stream>>>(h0, W_dec, decproj);

  const size_t attn_lds = (size_t)AMT * HH * sizeof(bf16_t) + AMT * sizeof(float);
  attn_score_kernel<<<dim3(SS / AMT, BB), 256, attn_lds, stream>>>(
      enc, W_enc, decproj, w_val, scores);

  softmax_kernel<<<BB, 256, 0, stream>>>(scores, out_attn);

  context_kernel<<<dim3(HH / 256, BB), 256, 0, stream>>>(enc, out_attn, context);

  sim_kernel<<<BB, 64, 0, stream>>>(context, events, W_gate, out_sim);

  gates_kernel<<<N4H / 128, 256, 0, stream>>>(
      context, h0, out_sim, x, W_ih, W_hh, b_ih, b_hh, gates);

  lstm_ln_kernel<<<BB, 256, 0, stream>>>(
      gates, c0, ln_g, ln_b, W_fin, b_fin, out_fin, out_h, out_c);
}
